// String_Finder_30897994728003
// MI455X (gfx1250) — compile-verified, matrix operand order checked by reading, not run on hardware
//
#include <hip/hip_runtime.h>

typedef __attribute__((ext_vector_type(16))) _Float16 v16h;
typedef __attribute__((ext_vector_type(8)))  float    v8f;

#define IMG     64
#define NPIX    4096        // 64*64 queries
#define NEB     841
#define KSTEPS  27          // 864 = 27 * 32 (K padded)
#define NDB     16384
#define MTILES  256         // 4096 / 16
#define NTILES  1024        // 16384 / 16
#define TOPK    40
#define WAVES   4
#define NBLK    4           // N-tiles per wave iteration (A-fragment reuse factor)
#define NGROUPS (NTILES / NBLK)          // 256
#define GRP_PER_WAVE (NGROUPS / WAVES)   // 64
#define FINF    3.4e38f

// workspace byte offsets (all 32B-aligned)
#define OFF_SEED 0u            // _Float16[4096]
#define OFF_Q2   8192u         // float[4096]
#define OFF_D2   24576u        // float[16384]
#define OFF_AF   90112u        // _Float16[256*27*512]  (~7.1 MB)
#define OFF_BF   7168000u      // _Float16[1024*27*512] (~28.3 MB)

// ---------------- seed binarization ----------------
__global__ void k_seed(const float* __restrict__ batch, _Float16* __restrict__ seed) {
    int i = blockIdx.x * blockDim.x + threadIdx.x;
    if (i >= NPIX) return;
    int y = i >> 6;
    int d = y - (IMG / 2);
    bool zeroed = (d > -4) && (d < 6);
    float v = zeroed ? 0.0f : batch[i];
    seed[i] = (v > 0.7f) ? (_Float16)1.0f : (_Float16)0.0f;
}

// ---------------- q2 = |rel|^2 (binary -> count of ones) ----------------
__global__ void k_q2(const _Float16* __restrict__ seed, float* __restrict__ q2) {
    int q = blockIdx.x * blockDim.x + threadIdx.x;
    if (q >= NPIX) return;
    int y0 = (q >> 6) - 14, x0 = (q & 63) - 14;
    float s = 0.0f;
    for (int dy = 0; dy < 29; ++dy) {
        int y = y0 + dy;
        if ((unsigned)y >= 64u) continue;
        for (int dx = 0; dx < 29; ++dx) {
            int x = x0 + dx;
            if ((unsigned)x >= 64u) continue;
            s += (float)seed[y * 64 + x];
        }
    }
    q2[q] = s;
}

// ---------------- d2 = |db_n|^2 ----------------
__global__ void k_d2(const float* __restrict__ db, float* __restrict__ d2) {
    int n = blockIdx.x * blockDim.x + threadIdx.x;
    if (n >= NDB) return;
    const float* row = db + (size_t)n * NEB;
    float s = 0.0f;
    for (int j = 0; j < NEB; ++j) { float v = row[j]; s += v * v; }
    d2[n] = s;
}

// K index inside a 16x32 f16 WMMA fragment: lane l, element e (of v16h)
__device__ __forceinline__ int frag_k(int lane, int e) {
    return ((e >> 3) << 4) + ((lane >> 4) << 3) + (e & 7);
}

// ---------------- pack A (queries) into WMMA fragment order ----------------
// grid.x = MTILES*KSTEPS, block = 512 (one thread per fragment half)
__global__ void k_packA(const _Float16* __restrict__ seed, _Float16* __restrict__ Af) {
    int blk = blockIdx.x;
    int kstep = blk % KSTEPS;
    int t = threadIdx.x;            // t = lane*16 + e
    int lane = t >> 4, e = t & 15;
    int q = (blk / KSTEPS) * 16 + (lane & 15);
    int K = kstep * 32 + frag_k(lane, e);
    _Float16 v = (_Float16)0.0f;
    if (K < NEB) {
        int dy = K / 29, dx = K % 29;
        int y = (q >> 6) + dy - 14;
        int x = (q & 63) + dx - 14;
        if ((unsigned)y < 64u && (unsigned)x < 64u) v = seed[y * 64 + x];
    }
    Af[(size_t)blk * 512 + t] = v;
}

// ---------------- pack B (db) into WMMA fragment order ----------------
// grid.x = NTILES*KSTEPS, block = 512
__global__ void k_packB(const float* __restrict__ db, _Float16* __restrict__ Bf) {
    int blk = blockIdx.x;
    int kstep = blk % KSTEPS;
    int t = threadIdx.x;
    int lane = t >> 4, e = t & 15;
    int n = (blk / KSTEPS) * 16 + (lane & 15);
    int K = kstep * 32 + frag_k(lane, e);
    _Float16 v = (_Float16)0.0f;
    if (K < NEB) v = (_Float16)db[(size_t)n * NEB + K];
    Bf[(size_t)blk * 512 + t] = v;
}

// ---------------- fused WMMA GEMM + squared distance + top-40 ----------------
// grid.x = MTILES, block = 128 (4 waves); each wave computes 16x64 per iteration
__global__ void __launch_bounds__(128)
k_gemm_topk(const _Float16* __restrict__ Af, const _Float16* __restrict__ Bf,
            const float* __restrict__ q2, const float* __restrict__ d2,
            float* __restrict__ out_d, float* __restrict__ out_i)
{
    __shared__ __align__(32) _Float16 Alds[KSTEPS * 512];   // 27648 B
    __shared__ float stage[WAVES * NBLK * 256];             // 16384 B
    __shared__ float listd[WAVES * 16 * TOPK];              // 10240 B
    __shared__ int   listi[WAVES * 16 * TOPK];              // 10240 B

    const int m_tile = blockIdx.x;
    const int tid  = threadIdx.x;
    const int lane = tid & 31;
    const int wave = tid >> 5;

    // stage A fragments for this query tile into LDS
    const _Float16* Asrc = Af + (size_t)m_tile * KSTEPS * 512;
    for (int i = tid; i < KSTEPS * 512; i += 128) Alds[i] = Asrc[i];
    for (int i = tid; i < WAVES * 16 * TOPK; i += 128) { listd[i] = FINF; listi[i] = 0; }
    __syncthreads();

    // q2 for the 8 rows this lane's accumulators cover: M = r + 8*(lane>>4)
    float q2v[8];
    {
        int mbase = m_tile * 16 + ((lane >> 4) << 3);
        for (int r = 0; r < 8; ++r) q2v[r] = q2[mbase + r];
    }

    float curmax = FINF; int argmax = 0;                    // owner-lane register state
    float* myd = &listd[(wave * 16 + (lane & 15)) * TOPK];
    int*   myi = &listi[(wave * 16 + (lane & 15)) * TOPK];

    const int mrow = (lane >> 4) << 3;
    const int col  = lane & 15;

    for (int it = 0; it < GRP_PER_WAVE; ++it) {
        const int n_tile0 = (wave + it * WAVES) * NBLK;

        // prefetch next group's B stream into L2 (global_prefetch_b8)
        if (it + 1 < GRP_PER_WAVE) {
            const int nnext = (wave + (it + 1) * WAVES) * NBLK;
            __builtin_prefetch(Bf + (size_t)nnext * KSTEPS * 512, 0, 1);
        }

        v8f acc[NBLK];
        #pragma unroll
        for (int j = 0; j < NBLK; ++j) acc[j] = (v8f){};

        const v16h* Bp[NBLK];
        #pragma unroll
        for (int j = 0; j < NBLK; ++j)
            Bp[j] = (const v16h*)(Bf + (size_t)(n_tile0 + j) * KSTEPS * 512);

        for (int k = 0; k < KSTEPS; ++k) {
            v16h a = *(const v16h*)(&Alds[k * 512 + lane * 16]);   // ds_load_b128 x2, shared by 4 WMMAs
            #pragma unroll
            for (int j = 0; j < NBLK; ++j) {
                v16h b = Bp[j][k * 32 + lane];                     // global_load_b128 x2 (L2-resident)
                acc[j] = __builtin_amdgcn_wmma_f32_16x16x32_f16(
                             false, a, false, b, (short)0, acc[j], false, false);
            }
        }

        float d2n[NBLK];
        #pragma unroll
        for (int j = 0; j < NBLK; ++j) d2n[j] = d2[(n_tile0 + j) * 16 + col];

        __syncthreads();   // previous iteration's readers done before overwrite
        #pragma unroll
        for (int j = 0; j < NBLK; ++j) {
            float* st = &stage[(wave * NBLK + j) * 256];
            #pragma unroll
            for (int r = 0; r < 8; ++r)
                st[(mrow + r) * 16 + col] = q2v[r] - 2.0f * acc[j][r] + d2n[j];
        }
        __syncthreads();

        if (lane < 16) {   // lane owns query row `lane` of this tile
            for (int j = 0; j < NBLK; ++j) {
                const float* st = &stage[(wave * NBLK + j) * 256 + lane * 16];
                int nbase = (n_tile0 + j) * 16;
                for (int c = 0; c < 16; ++c) {
                    float v = st[c];
                    if (v < curmax) {
                        myd[argmax] = v;
                        myi[argmax] = nbase + c;
                        float m = -1.0f; int am = 0;
                        for (int t2 = 0; t2 < TOPK; ++t2) {
                            float dv = myd[t2];
                            if (dv > m) { m = dv; am = t2; }
                        }
                        curmax = m; argmax = am;
                    }
                }
            }
        }
    }
    __syncthreads();

    // merge the 4 per-wave lists per query; emit sorted ascending (matches top_k)
    if (tid < 16) {
        int q  = tid;
        int qg = m_tile * 16 + q;
        for (int k = 0; k < TOPK; ++k) {
            float best = FINF; int bw = 0, bt = 0;
            for (int w = 0; w < WAVES; ++w) {
                const float* ld = &listd[(w * 16 + q) * TOPK];
                for (int t2 = 0; t2 < TOPK; ++t2) {
                    float v = ld[t2];
                    if (v < best) { best = v; bw = w; bt = t2; }
                }
            }
            int bi = listi[(bw * 16 + q) * TOPK + bt];
            listd[(bw * 16 + q) * TOPK + bt] = FINF;
            out_d[qg * TOPK + k] = best;
            out_i[qg * TOPK + k] = (float)bi;
        }
    }
}

extern "C" void kernel_launch(void* const* d_in, const int* in_sizes, int n_in,
                              void* d_out, int out_size, void* d_ws, size_t ws_size,
                              hipStream_t stream) {
    (void)in_sizes; (void)n_in; (void)out_size; (void)ws_size;
    const float* batch = (const float*)d_in[0];   // [1,1,64,64]
    const float* db    = (const float*)d_in[1];   // [16384,841]

    char* ws = (char*)d_ws;
    _Float16* seed = (_Float16*)(ws + OFF_SEED);
    float*    q2   = (float*)(ws + OFF_Q2);
    float*    d2   = (float*)(ws + OFF_D2);
    _Float16* Af   = (_Float16*)(ws + OFF_AF);
    _Float16* Bf   = (_Float16*)(ws + OFF_BF);

    float* out = (float*)d_out;                   // dists [4096*40] ++ idx-as-float [4096*40]

    k_seed <<<NPIX / 256, 256, 0, stream>>>(batch, seed);
    k_q2   <<<NPIX / 256, 256, 0, stream>>>(seed, q2);
    k_d2   <<<NDB / 256, 256, 0, stream>>>(db, d2);
    k_packA<<<MTILES * KSTEPS, 512, 0, stream>>>(seed, Af);
    k_packB<<<NTILES * KSTEPS, 512, 0, stream>>>(db, Bf);
    k_gemm_topk<<<MTILES, 128, 0, stream>>>(Af, Bf, q2, d2, out, out + NPIX * TOPK);
}
